// DynamicGCN_66374424592407
// MI455X (gfx1250) — matrix operand order repaired and checked by reading, not compiled, not run on hardware
//
#include <hip/hip_runtime.h>
#include <hip/hip_bf16.h>
#include <math.h>

// ---------------------------------------------------------------------------
// DynamicGCN on MI455X (gfx1250, wave32, WMMA).
// Dense GEMMs -> v_wmma_f32_16x16x32_bf16 (fp32 accum).
//   A operand: direct per-lane global_load_b128 from L2-resident activations,
//              packed f32->bf16 conversion in registers (no LDS, no barriers).
//   B operand: weights staged into LDS in *fragment-major* layout so a
//              fragment read is 2x ds_load_b128; column tiles processed in
//              pairs with distinct fragment registers so the scheduler can
//              overlap LDS latency with WMMA issue (partial dscnt waits).
//   Dual M-tile per wave: each B fragment feeds two WMMAs (independent
//              accumulator chains), halving LDS traffic per matrix op.
// SpMM -> L2-resident gather + fp32 global atomics.
// BatchNorm -> two-pass with per-feature atomics.
// Temporal gate -> fused dual-GEMM + sigmoid + in-place blend.
// ---------------------------------------------------------------------------

typedef __attribute__((ext_vector_type(16))) __bf16 v16bf;
typedef __attribute__((ext_vector_type(8)))  float  v8f;

#define NLAYER 7
#define NHID   6

__device__ __forceinline__ void atomic_add_f32(float* p, float v) {
  __hip_atomic_fetch_add(p, v, __ATOMIC_RELAXED, __HIP_MEMORY_SCOPE_AGENT);
}

__device__ __forceinline__ float sigmoidf_fast(float z) {
  return 1.0f / (1.0f + __expf(-z));
}

// Build 16x32 bf16 A-fragment for this lane from 16 fp32 values:
// j=0..7  -> A[row][kb+0..7],  j=8..15 -> A[row][kb+16..23]  (ISA 7.12.2)
__device__ __forceinline__ v16bf make_afrag(const float4& a0, const float4& a1,
                                            const float4& a2, const float4& a3) {
  v16bf f;
  f[0]  = (__bf16)a0.x; f[1]  = (__bf16)a0.y; f[2]  = (__bf16)a0.z; f[3]  = (__bf16)a0.w;
  f[4]  = (__bf16)a1.x; f[5]  = (__bf16)a1.y; f[6]  = (__bf16)a1.z; f[7]  = (__bf16)a1.w;
  f[8]  = (__bf16)a2.x; f[9]  = (__bf16)a2.y; f[10] = (__bf16)a2.z; f[11] = (__bf16)a2.w;
  f[12] = (__bf16)a3.x; f[13] = (__bf16)a3.y; f[14] = (__bf16)a3.z; f[15] = (__bf16)a3.w;
  return f;
}

// Load this lane's A fragment (rows of fp32 matrix with ld=128) for K-chunk
// starting at koff; zero-filled if row out of range.
__device__ __forceinline__ v16bf load_afrag(const float* __restrict__ base,
                                            int row, int nrows, int koff) {
  float4 a0 = make_float4(0.f, 0.f, 0.f, 0.f), a1 = a0, a2 = a0, a3 = a0;
  if (row < nrows) {
    const float* ap = base + (size_t)row * 128 + koff;
    a0 = *(const float4*)(ap + 0);
    a1 = *(const float4*)(ap + 4);
    a2 = *(const float4*)(ap + 16);
    a3 = *(const float4*)(ap + 20);
  }
  return make_afrag(a0, a1, a2, a3);
}

// ---------------- gather ----------------------------------------------------
__global__ void k_gather(const int* __restrict__ verts,
                         const float* __restrict__ emb,
                         float* __restrict__ X, int nrows) {
  long gid = (long)blockIdx.x * blockDim.x + threadIdx.x;
  if (gid >= (long)nrows * 128) return;
  int n = (int)(gid >> 7);
  int d = (int)(gid & 127);
  X[gid] = emb[(size_t)verts[n] * 128 + d];
}

__global__ void k_gather_mw(const int* __restrict__ verts,
                            const float* __restrict__ maskw,
                            float* __restrict__ mw, int nrows) {
  int n = blockIdx.x * blockDim.x + threadIdx.x;
  if (n < nrows) mw[n] = maskw[verts[n]];
}

// ---------------- small utility kernels ------------------------------------
__global__ void k_zero(float* __restrict__ p, int n) {
  int i = blockIdx.x * blockDim.x + threadIdx.x;
  if (i < n) p[i] = 0.0f;
}

__global__ void k_init_bias(float* __restrict__ H, const float* __restrict__ bias,
                            int nrows, int D) {
  long gid = (long)blockIdx.x * blockDim.x + threadIdx.x;
  if (gid >= (long)nrows * D) return;
  H[gid] = bias[gid % D];
}

// ---------------- WMMA GEMM: C[n x ncols] = A[n x 128] @ W[128 x ncols] -----
// 128 threads = 4 waves; wave w owns rows [w*32, w*32+32) as two 16-row
// M-tiles. Block covers the full output width with COLT 16-wide column tiles
// (COLT even). Whole W staged once in LDS, fragment-major.
template <int COLT>
__global__ __launch_bounds__(128) void k_gemm_wmma(
    const float* __restrict__ A, const float* __restrict__ W,
    float* __restrict__ C, int nrows, int ncols, int ldw, int ldc) {
  constexpr int WB = COLT * 16;
  __shared__ __bf16 frB[4][COLT][32][16];   // 32KB for COLT=8

  const int tid  = threadIdx.x;
  const int lane = tid & 31;
  const int wave = tid >> 5;                // 0..3
  const int rowBase = blockIdx.x * 128;
  const int waveRow = wave * 32;

  // ---- stage all of W (128 x ncols) into fragment-major LDS, k-pairs ----
  for (int p = tid; p < 64 * WB; p += 128) {
    int kk = p / WB;              // k-pair index 0..63
    int c  = p % WB;
    int k  = kk * 2;
    float v0 = (c < ncols) ? W[(size_t)k * ldw + c] : 0.0f;
    float v1 = (c < ncols) ? W[(size_t)(k + 1) * ldw + c] : 0.0f;
    int chunk = k >> 5;
    int kl    = k & 31;
    int bl    = (c & 15) + (((kl >> 3) & 1) << 4);   // fragment lane
    int ct    = c >> 4;
    int j     = (kl & 7) + ((kl >> 4) << 3);   // even -> 4B aligned pair
    __bf16* dp = &frB[chunk][ct][bl][j];
    dp[0] = (__bf16)v0;
    dp[1] = (__bf16)v1;
  }
  __syncthreads();

  v8f acc0[COLT], acc1[COLT];
#pragma unroll
  for (int t = 0; t < COLT; ++t)
#pragma unroll
    for (int r = 0; r < 8; ++r) { acc0[t][r] = 0.0f; acc1[t][r] = 0.0f; }

  const int r0  = rowBase + waveRow + (lane & 15);
  const int r1  = r0 + 16;
  const int kbA = (lane >> 4) * 8;             // A-frag K base

#pragma unroll
  for (int kc = 0; kc < 4; ++kc) {
    v16bf af0 = load_afrag(A, r0, nrows, kc * 32 + kbA);
    v16bf af1 = load_afrag(A, r1, nrows, kc * 32 + kbA);
#pragma unroll
    for (int ct = 0; ct < COLT; ct += 2) {
      // two independent fragments in distinct registers -> scheduler can
      // wait partially (dscnt<=2) before the first WMMA pair
      v16bf bA = *(const v16bf*)(&frB[kc][ct + 0][lane][0]);
      v16bf bB = *(const v16bf*)(&frB[kc][ct + 1][lane][0]);
      acc0[ct + 0] = __builtin_amdgcn_wmma_f32_16x16x32_bf16(
          false, af0, false, bA, (short)0, acc0[ct + 0], false, false);
      acc1[ct + 0] = __builtin_amdgcn_wmma_f32_16x16x32_bf16(
          false, af1, false, bA, (short)0, acc1[ct + 0], false, false);
      acc0[ct + 1] = __builtin_amdgcn_wmma_f32_16x16x32_bf16(
          false, af0, false, bB, (short)0, acc0[ct + 1], false, false);
      acc1[ct + 1] = __builtin_amdgcn_wmma_f32_16x16x32_bf16(
          false, af1, false, bB, (short)0, acc1[ct + 1], false, false);
    }
  }

  // store (C/D layout: lanes 0-15 -> M=vgpr, lanes 16-31 -> M=vgpr+8)
  const int nc   = lane & 15;
  const int mofs = (lane >> 4) * 8;
#pragma unroll
  for (int ct = 0; ct < COLT; ++ct) {
    int col = ct * 16 + nc;
    if (col >= ncols) continue;
#pragma unroll
    for (int r = 0; r < 8; ++r) {
      int orow = rowBase + waveRow + mofs + r;
      if (orow < nrows) C[(size_t)orow * ldc + col] = acc0[ct][r];
      int orow1 = orow + 16;
      if (orow1 < nrows) C[(size_t)orow1 * ldc + col] = acc1[ct][r];
    }
  }
}

// ---------------- fused temporal gate ---------------------------------------
// G = sigmoid(X@Wl + H@Wc + b);  X <- G*H + (1-G)*X.  In place: each wave's
// 32-row stripe is read (A operand + epilogue) and written only by itself.
__global__ __launch_bounds__(128) void k_gate_wmma(
    float* __restrict__ X, const float* __restrict__ H,
    const float* __restrict__ Wl, const float* __restrict__ Wc,
    const float* __restrict__ tb, int nrows) {
  __shared__ __bf16 frBl[8][32][16];   // 8KB  (one K-chunk of Wl)
  __shared__ __bf16 frBc[8][32][16];   // 8KB  (one K-chunk of Wc)

  const int tid  = threadIdx.x;
  const int lane = tid & 31;
  const int wave = tid >> 5;           // 0..3
  const int rowBase = blockIdx.x * 128;
  const int waveRow = wave * 32;

  v8f acc0[8], acc1[8];
#pragma unroll
  for (int t = 0; t < 8; ++t)
#pragma unroll
    for (int r = 0; r < 8; ++r) { acc0[t][r] = 0.0f; acc1[t][r] = 0.0f; }

  const int r0  = rowBase + waveRow + (lane & 15);
  const int r1  = r0 + 16;
  const int kbA = (lane >> 4) * 8;

  for (int kc = 0; kc < 4; ++kc) {
    __syncthreads();
    // stage one 32 x 128 chunk of each weight, fragment-major, k-pairs
    for (int p = tid; p < 16 * 128; p += 128) {
      int kk = p >> 7;                // k-pair 0..15 within chunk
      int c  = p & 127;
      int kl = kk * 2;                // local k, even
      int kg = kc * 32 + kl;          // global k
      int bl = (c & 15) + (((kl >> 3) & 1) << 4);
      int ct = c >> 4;
      int j  = (kl & 7) + ((kl >> 4) << 3);
      float l0 = Wl[(size_t)kg * 128 + c];
      float l1 = Wl[(size_t)(kg + 1) * 128 + c];
      float c0 = Wc[(size_t)kg * 128 + c];
      float c1 = Wc[(size_t)(kg + 1) * 128 + c];
      __bf16* dl = &frBl[ct][bl][j];
      dl[0] = (__bf16)l0; dl[1] = (__bf16)l1;
      __bf16* dc = &frBc[ct][bl][j];
      dc[0] = (__bf16)c0; dc[1] = (__bf16)c1;
    }
    __syncthreads();

    v16bf ax0 = load_afrag(X, r0, nrows, kc * 32 + kbA);
    v16bf ax1 = load_afrag(X, r1, nrows, kc * 32 + kbA);
    v16bf ah0 = load_afrag(H, r0, nrows, kc * 32 + kbA);
    v16bf ah1 = load_afrag(H, r1, nrows, kc * 32 + kbA);

#pragma unroll
    for (int ct = 0; ct < 8; ct += 2) {
      v16bf blA = *(const v16bf*)(&frBl[ct + 0][lane][0]);
      v16bf bcA = *(const v16bf*)(&frBc[ct + 0][lane][0]);
      v16bf blB = *(const v16bf*)(&frBl[ct + 1][lane][0]);
      v16bf bcB = *(const v16bf*)(&frBc[ct + 1][lane][0]);
      acc0[ct + 0] = __builtin_amdgcn_wmma_f32_16x16x32_bf16(
          false, ax0, false, blA, (short)0, acc0[ct + 0], false, false);
      acc1[ct + 0] = __builtin_amdgcn_wmma_f32_16x16x32_bf16(
          false, ax1, false, blA, (short)0, acc1[ct + 0], false, false);
      acc0[ct + 0] = __builtin_amdgcn_wmma_f32_16x16x32_bf16(
          false, ah0, false, bcA, (short)0, acc0[ct + 0], false, false);
      acc1[ct + 0] = __builtin_amdgcn_wmma_f32_16x16x32_bf16(
          false, ah1, false, bcA, (short)0, acc1[ct + 0], false, false);
      acc0[ct + 1] = __builtin_amdgcn_wmma_f32_16x16x32_bf16(
          false, ax0, false, blB, (short)0, acc0[ct + 1], false, false);
      acc1[ct + 1] = __builtin_amdgcn_wmma_f32_16x16x32_bf16(
          false, ax1, false, blB, (short)0, acc1[ct + 1], false, false);
      acc0[ct + 1] = __builtin_amdgcn_wmma_f32_16x16x32_bf16(
          false, ah0, false, bcB, (short)0, acc0[ct + 1], false, false);
      acc1[ct + 1] = __builtin_amdgcn_wmma_f32_16x16x32_bf16(
          false, ah1, false, bcB, (short)0, acc1[ct + 1], false, false);
    }
  }

  const int nc   = lane & 15;
  const int mofs = (lane >> 4) * 8;
#pragma unroll
  for (int ct = 0; ct < 8; ++ct) {
    int col = ct * 16 + nc;
    float bias = tb[col];
#pragma unroll
    for (int r = 0; r < 8; ++r) {
      int orow = rowBase + waveRow + mofs + r;
      if (orow < nrows) {
        size_t o = (size_t)orow * 128 + col;
        float g  = sigmoidf_fast(acc0[ct][r] + bias);
        X[o] = g * H[o] + (1.0f - g) * X[o];
      }
      int orow1 = orow + 16;
      if (orow1 < nrows) {
        size_t o = (size_t)orow1 * 128 + col;
        float g  = sigmoidf_fast(acc1[ct][r] + bias);
        X[o] = g * H[o] + (1.0f - g) * X[o];
      }
    }
  }
}

// ---------------- SpMM scatter-add ------------------------------------------
// H[dst[e], :] += w[e] * S[src[e], :]. 32 lanes/edge; D=128: 4 feats/lane.
// S (25.6 MB) is L2-resident on the 192 MB L2 -> gathers hit cache.
__global__ void k_spmm128(const int* __restrict__ src, const int* __restrict__ dst,
                          const float* __restrict__ w, const float* __restrict__ S,
                          float* __restrict__ H, int E) {
  int gid = blockIdx.x * blockDim.x + threadIdx.x;
  int e = gid >> 5;
  if (e >= E) return;
  int l = gid & 31;
  int s = src[e], d = dst[e];
  float wt = w[e];
  float4 v = *(const float4*)(S + (size_t)s * 128 + l * 4);
  float* o = H + (size_t)d * 128 + l * 4;
  atomic_add_f32(o + 0, wt * v.x);
  atomic_add_f32(o + 1, wt * v.y);
  atomic_add_f32(o + 2, wt * v.z);
  atomic_add_f32(o + 3, wt * v.w);
}

__global__ void k_spmm20(const int* __restrict__ src, const int* __restrict__ dst,
                         const float* __restrict__ w, const float* __restrict__ S,
                         float* __restrict__ H, int E) {
  int gid = blockIdx.x * blockDim.x + threadIdx.x;
  int e = gid >> 5;
  if (e >= E) return;
  int l = gid & 31;
  if (l >= 20) return;
  int s = src[e], d = dst[e];
  float wt = w[e];
  atomic_add_f32(H + (size_t)d * 20 + l, wt * S[(size_t)s * 20 + l]);
}

// ---------------- BatchNorm (training-mode batch stats) ---------------------
__global__ void k_bn_stats(const float* __restrict__ H, float* __restrict__ sum,
                           float* __restrict__ sq, int nrows, int D, int rowsPerBlock) {
  int f = threadIdx.x;
  if (f >= D) return;
  int r0 = blockIdx.x * rowsPerBlock;
  int r1 = r0 + rowsPerBlock;
  if (r1 > nrows) r1 = nrows;
  float s = 0.f, q = 0.f;
  for (int r = r0; r < r1; ++r) {
    float v = H[(size_t)r * D + f];
    s += v;
    q += v * v;
  }
  atomic_add_f32(sum + f, s);
  atomic_add_f32(sq + f, q);
}

__global__ void k_bn_apply(float* __restrict__ H, const float* __restrict__ sum,
                           const float* __restrict__ sq, const float* __restrict__ gamma,
                           const float* __restrict__ beta, int nrows, int D, int doRelu) {
  long gid = (long)blockIdx.x * blockDim.x + threadIdx.x;
  if (gid >= (long)nrows * D) return;
  int f = (int)(gid % D);
  float invn = 1.0f / (float)nrows;
  float mean = sum[f] * invn;
  float var  = sq[f] * invn - mean * mean;
  if (var < 0.f) var = 0.f;
  float v = (H[gid] - mean) * rsqrtf(var + 1e-5f) * gamma[f] + beta[f];
  if (doRelu) v = fmaxf(v, 0.0f);
  H[gid] = v;
}

// ---------------- final masked dot reduction --------------------------------
// accum[o] = sum_n mw[n] * relu(X20[n,o])
__global__ __launch_bounds__(256) void k_maskdot(const float* __restrict__ X20,
                                                 const float* __restrict__ mw,
                                                 float* __restrict__ accum, int nrows) {
  __shared__ float red[256];
  float a[20];
#pragma unroll
  for (int o = 0; o < 20; ++o) a[o] = 0.0f;
  for (int n = blockIdx.x * 256 + threadIdx.x; n < nrows; n += gridDim.x * 256) {
    float m = mw[n];
#pragma unroll
    for (int o = 0; o < 20; ++o)
      a[o] += m * fmaxf(X20[(size_t)n * 20 + o], 0.0f);
  }
#pragma unroll
  for (int o = 0; o < 20; ++o) {
    red[threadIdx.x] = a[o];
    __syncthreads();
    for (int s = 128; s > 0; s >>= 1) {
      if (threadIdx.x < s) red[threadIdx.x] += red[threadIdx.x + s];
      __syncthreads();
    }
    if (threadIdx.x == 0) atomic_add_f32(accum + o, red[0]);
    __syncthreads();
  }
}

__global__ void k_final(const float* __restrict__ accum, const float* __restrict__ maskb,
                        float* __restrict__ out) {
  int o = threadIdx.x;
  if (o < 20) out[o] = sigmoidf_fast(accum[o] + maskb[o]);
}

// ---------------------------------------------------------------------------
extern "C" void kernel_launch(void* const* d_in, const int* in_sizes, int n_in,
                              void* d_out, int out_size, void* d_ws, size_t ws_size,
                              hipStream_t stream) {
  (void)n_in; (void)out_size; (void)ws_size;
  const int*   verts = (const int*)d_in[0];
  const int*   eidx  = (const int*)d_in[1];
  const float* ew    = (const float*)d_in[2];
  const float* emb   = (const float*)d_in[3];
  const float* wh    = (const float*)d_in[4];
  const float* bh    = (const float*)d_in[5];
  const float* wout  = (const float*)d_in[6];
  const float* bout  = (const float*)d_in[7];
  const float* bng_h = (const float*)d_in[8];
  const float* bnb_h = (const float*)d_in[9];
  const float* bng_o = (const float*)d_in[10];
  const float* bnb_o = (const float*)d_in[11];
  const float* tewl  = (const float*)d_in[12];
  const float* tewc  = (const float*)d_in[13];
  const float* teb   = (const float*)d_in[14];
  const float* maskw = (const float*)d_in[15];
  const float* maskb = (const float*)d_in[16];
  float* out = (float*)d_out;

  const int N = in_sizes[0];
  const int E = in_sizes[2] / NLAYER;

  // workspace carve-out (256B aligned slices)
  char* ws = (char*)d_ws;
  size_t off = 0;
  auto carve = [&](size_t bytes) -> float* {
    float* p = (float*)(ws + off);
    off += (bytes + 255) & ~(size_t)255;
    return p;
  };
  float* X     = carve((size_t)N * 128 * 4);
  float* S     = carve((size_t)N * 128 * 4);
  float* H     = carve((size_t)N * 128 * 4);
  float* S20   = carve((size_t)N * 20 * 4);
  float* H20   = carve((size_t)N * 20 * 4);
  float* MW    = carve((size_t)N * 4);
  float* STATS = carve(256 * 4);          // [0:128)=sum, [128:256)=sumsq
  float* ACC   = carve(32 * 4);

  const long totalX  = (long)N * 128;
  const long total20 = (long)N * 20;
  const int  gemmGrid = (N + 127) / 128;
  const int  spGrid   = (int)(((long)E * 32 + 255) / 256);

  k_gather<<<(int)((totalX + 255) / 256), 256, 0, stream>>>(verts, emb, X, N);
  k_gather_mw<<<(N + 255) / 256, 256, 0, stream>>>(verts, maskw, MW, N);

  for (int i = 0; i < NHID; ++i) {
    // support = X @ W_h[i]
    k_gemm_wmma<8><<<gemmGrid, 128, 0, stream>>>(
        X, wh + (size_t)i * 128 * 128, S, N, 128, 128, 128);
    // h = spmm(...) + b   (init with bias, then scatter-add)
    k_init_bias<<<(int)((totalX + 255) / 256), 256, 0, stream>>>(H, bh + i * 128, N, 128);
    const int* src = eidx + (size_t)i * 2 * E;
    const int* dst = src + E;
    k_spmm128<<<spGrid, 256, 0, stream>>>(src, dst, ew + (size_t)i * E, S, H, E);
    // batchnorm + relu
    k_zero<<<1, 256, 0, stream>>>(STATS, 256);
    k_bn_stats<<<(N + 63) / 64, 128, 0, stream>>>(H, STATS, STATS + 128, N, 128, 64);
    k_bn_apply<<<(int)((totalX + 255) / 256), 256, 0, stream>>>(
        H, STATS, STATS + 128, bng_h + i * 128, bnb_h + i * 128, N, 128, 1);
    // temporal gate: X <- g*H + (1-g)*X, g = sigmoid(X@Wl[j] + H@Wc[j] + b[j])
    int jj = (i + NHID - 1) % NHID;  // Python (i-1) % 6 semantics
    k_gate_wmma<<<gemmGrid, 128, 0, stream>>>(
        X, H, tewl + (size_t)jj * 128 * 128, tewc + (size_t)jj * 128 * 128,
        teb + jj * 128, N);
  }

  // output layer
  k_gemm_wmma<2><<<gemmGrid, 128, 0, stream>>>(X, wout, S20, N, 20, 20, 20);
  k_init_bias<<<(int)((total20 + 255) / 256), 256, 0, stream>>>(H20, bout, N, 20);
  {
    const int* src = eidx + (size_t)6 * 2 * E;
    const int* dst = src + E;
    k_spmm20<<<spGrid, 256, 0, stream>>>(src, dst, ew + (size_t)6 * E, S20, H20, E);
  }
  k_zero<<<1, 256, 0, stream>>>(STATS, 256);
  k_bn_stats<<<(N + 63) / 64, 32, 0, stream>>>(H20, STATS, STATS + 128, N, 20, 64);
  k_bn_apply<<<(int)((total20 + 255) / 256), 256, 0, stream>>>(
      H20, STATS, STATS + 128, bng_o, bnb_o, N, 20, 0);

  // final relu + masked dot + sigmoid
  k_zero<<<1, 32, 0, stream>>>(ACC, 32);
  k_maskdot<<<128, 256, 0, stream>>>(H20, MW, ACC, N);
  k_final<<<1, 32, 0, stream>>>(ACC, maskb, out);
}
